// SHRED_79525614452818
// MI455X (gfx1250) — matrix operand order repaired and checked by reading, not compiled, not run on hardware
//
#include <hip/hip_runtime.h>
#include <hip/hip_bf16.h>

typedef __attribute__((ext_vector_type(16))) _Float16 v16h;
typedef __attribute__((ext_vector_type(8)))  _Float16 v8h;
typedef __attribute__((ext_vector_type(8)))  float    v8f;

union AF { v16h v; v8h h[2]; };

#define WMMA_F16(a, b, c) \
  __builtin_amdgcn_wmma_f32_16x16x32_f16(false, (a), false, (b), (short)0, (c), false, false)

// A fragment (16x32 f16, M x K): lane holds row M=lane&15; elems 0..15 at
// K = 2*hi8 .. 2*hi8+15 (hi8 = (lane>>4)*8).
__device__ __forceinline__ v16h fragA(const _Float16* p, int hi8) {
  AF f;
  f.h[0] = *(const v8h*)(p + 2 * hi8);
  f.h[1] = *(const v8h*)(p + 2 * hi8 + 8);
  return f.v;
}
// B fragment (32x16 f16, K x N) from row-major W[n][k]: lane holds col
// N=lane&15; elems 0..15 = K = hi16 .. hi16+15 (hi16 = (lane>>4)*16).
__device__ __forceinline__ v16h fragB(const _Float16* p, int hi16) {
  AF f;
  f.h[0] = *(const v8h*)(p + hi16);
  f.h[1] = *(const v8h*)(p + hi16 + 8);
  return f.v;
}

// Fast gate activations: 2 TRANS ops each (v_exp_f32 + v_rcp_f32), stable at
// +/-inf. These co-execute with the VALU/XDL pipes (ISA 8.4).
__device__ __forceinline__ float sigf(float x) {
  return __builtin_amdgcn_rcpf(1.0f + __expf(-x));
}
__device__ __forceinline__ float tanh_fast(float x) {
  float e = __expf(2.0f * x);                  // 0 .. inf
  return 1.0f - 2.0f * __builtin_amdgcn_rcpf(e + 1.0f);
}

// ---------------------------------------------------------------------------
// Persistent single-workgroup 2-layer LSTM over T=512 steps.
// 512 threads = 16 waves. Wave w: gate = w>>2, mt = w&3 -> owns output tile
// rows mt*16..+15, cols gate*64 + ntl*16 (ntl=0..3). All weights f16 in LDS.
// Cell state c lives in registers. x is double-buffered in LDS (prefetched one
// step ahead); dropout masks prefetched into registers -> global latency
// overlaps the layer-0 GEMM. 4 barriers per step.
// ---------------------------------------------------------------------------
__global__ __launch_bounds__(512) void lstm2_wmma_kernel(
    const float* __restrict__ x, const float* __restrict__ dm,
    const float* __restrict__ fm,
    const float* __restrict__ Wih0, const float* __restrict__ Whh0,
    const float* __restrict__ b0,
    const float* __restrict__ Wih1, const float* __restrict__ Whh1,
    const float* __restrict__ b1,
    float* __restrict__ hfin) {
  extern __shared__ __align__(16) char smem[];
  _Float16* sWih0 = (_Float16*)smem;        // [256][32]
  _Float16* sWhh0 = sWih0 + 256 * 32;       // [256][64]
  _Float16* sWih1 = sWhh0 + 256 * 64;       // [256][64]
  _Float16* sWhh1 = sWih1 + 256 * 64;       // [256][64]
  _Float16* sX    = sWhh1 + 256 * 64;       // [2][64][32] double buffer
  _Float16* sH0   = sX + 2 * 64 * 32;       // [64][64]
  _Float16* sH1   = sH0 + 64 * 64;          // [64][64]
  float* sB0 = (float*)(sH1 + 64 * 64);     // [256]
  float* sB1 = sB0 + 256;                   // [256]
  float* sZ  = sB1 + 256;                   // [64][256]

  const int tid = threadIdx.x;

  // One-time: convert weights f32 -> f16 into LDS, zero h state, stage x_0.
  for (int i = tid; i < 256 * 32; i += 512) sWih0[i] = (_Float16)Wih0[i];
  for (int i = tid; i < 256 * 64; i += 512) {
    sWhh0[i] = (_Float16)Whh0[i];
    sWih1[i] = (_Float16)Wih1[i];
    sWhh1[i] = (_Float16)Whh1[i];
  }
  if (tid < 256) { sB0[tid] = b0[tid]; sB1[tid] = b1[tid]; }
  for (int i = tid; i < 64 * 64; i += 512) {
    sH0[i] = (_Float16)0.f;
    sH1[i] = (_Float16)0.f;
  }
  for (int i = tid; i < 64 * 32; i += 512) {
    int bb = i >> 5, dd = i & 31;
    sX[i] = (_Float16)x[bb * (512 * 32) + dd];  // t = 0
  }
  __syncthreads();

  const int lane = tid & 31;
  const int wave = tid >> 5;
  const int gate = wave >> 2;
  const int mt   = wave & 3;
  const int m    = mt * 16 + (lane & 15);
  const int hi8  = (lane >> 4) * 8;
  const int hi16 = (lane >> 4) * 16;
  const int zrow = mt * 16 + (lane >> 4) * 8;

  // Elementwise ownership: 8 contiguous (b,h) elements per thread.
  const int eb  = (tid * 8) >> 6;   // batch row
  const int eh0 = (tid * 8) & 63;   // hidden base
  float c0r[8], c1r[8];
#pragma unroll
  for (int j = 0; j < 8; ++j) { c0r[j] = 0.f; c1r[j] = 0.f; }

  for (int t = 0; t < 512; ++t) {
    const _Float16* sXc = sX + (t & 1) * (64 * 32);
    _Float16* sXn       = sX + ((t + 1) & 1) * (64 * 32);

    // Prefetch dropout masks for this step (consumed after barriers) and
    // x_{t+1} into the other LDS buffer (consumed next iteration).
    float dm0r[8], dm1r[8];
    {
      const float* p0 = dm + ((t * 64 + eb) * 2 + 0) * 64 + eh0;
      const float* p1 = p0 + 64;
#pragma unroll
      for (int j = 0; j < 8; ++j) { dm0r[j] = p0[j]; dm1r[j] = p1[j]; }
    }
    if (t + 1 < 512) {
      for (int i = tid; i < 64 * 32; i += 512) {
        int bb = i >> 5, dd = i & 31;
        sXn[i] = (_Float16)x[bb * (512 * 32) + (t + 1) * 32 + dd];
      }
    }

    // ---- layer 0 GEMM: z = x_t @ Wih0^T + h0 @ Whh0^T  (f32 accum)
    {
      v16h aX  = fragA(sXc + m * 32, hi8);
      v16h aH0 = fragA(sH0 + m * 64, hi8);
      v16h aH1 = fragA(sH0 + m * 64 + 32, hi8);
#pragma unroll
      for (int ntl = 0; ntl < 4; ++ntl) {
        int n = gate * 64 + ntl * 16 + (lane & 15);
        v8f acc = {};
        acc = WMMA_F16(aX,  fragB(sWih0 + n * 32,      hi16), acc);
        acc = WMMA_F16(aH0, fragB(sWhh0 + n * 64,      hi16), acc);
        acc = WMMA_F16(aH1, fragB(sWhh0 + n * 64 + 32, hi16), acc);
#pragma unroll
        for (int r = 0; r < 8; ++r) sZ[(zrow + r) * 256 + n] = acc[r];
      }
    }
    __syncthreads();

    // ---- layer 0 gates + per-step dropout
    {
      const float* zb = sZ + eb * 256;
#pragma unroll
      for (int j = 0; j < 8; ++j) {
        int hh = eh0 + j;
        float zi = zb[hh]       + sB0[hh];
        float zf = zb[64 + hh]  + sB0[64 + hh];
        float zg = zb[128 + hh] + sB0[128 + hh];
        float zo = zb[192 + hh] + sB0[192 + hh];
        float cn = sigf(zf) * c0r[j] + sigf(zi) * tanh_fast(zg);
        c0r[j] = cn;
        float hn = sigf(zo) * tanh_fast(cn) * dm0r[j];
        sH0[eb * 64 + hh] = (_Float16)hn;
      }
    }
    __syncthreads();

    // ---- layer 1 GEMM: z = h0 @ Wih1^T + h1 @ Whh1^T
    {
      v16h aI0 = fragA(sH0 + m * 64, hi8);
      v16h aI1 = fragA(sH0 + m * 64 + 32, hi8);
      v16h aR0 = fragA(sH1 + m * 64, hi8);
      v16h aR1 = fragA(sH1 + m * 64 + 32, hi8);
#pragma unroll
      for (int ntl = 0; ntl < 4; ++ntl) {
        int n = gate * 64 + ntl * 16 + (lane & 15);
        v8f acc = {};
        acc = WMMA_F16(aI0, fragB(sWih1 + n * 64,      hi16), acc);
        acc = WMMA_F16(aI1, fragB(sWih1 + n * 64 + 32, hi16), acc);
        acc = WMMA_F16(aR0, fragB(sWhh1 + n * 64,      hi16), acc);
        acc = WMMA_F16(aR1, fragB(sWhh1 + n * 64 + 32, hi16), acc);
#pragma unroll
        for (int r = 0; r < 8; ++r) sZ[(zrow + r) * 256 + n] = acc[r];
      }
    }
    __syncthreads();

    // ---- layer 1 gates + dropout; final step also applies final_mask.
    {
      const float* zb = sZ + eb * 256;
#pragma unroll
      for (int j = 0; j < 8; ++j) {
        int hh = eh0 + j;
        float zi = zb[hh]       + sB1[hh];
        float zf = zb[64 + hh]  + sB1[64 + hh];
        float zg = zb[128 + hh] + sB1[128 + hh];
        float zo = zb[192 + hh] + sB1[192 + hh];
        float cn = sigf(zf) * c1r[j] + sigf(zi) * tanh_fast(zg);
        c1r[j] = cn;
        float hn = sigf(zo) * tanh_fast(cn) * dm1r[j];
        sH1[eb * 64 + hh] = (_Float16)hn;
        if (t == 511) hfin[eb * 64 + hh] = hn * fm[eb * 64 + hh];
      }
    }
    __syncthreads();
  }
}

// ---------------------------------------------------------------------------
// fc1 + fc2 (tiny: ~10 MFLOP). Single block, h1 staged in LDS; emits h2 as
// f16 padded to K=416 for the WMMA fc3 kernel.
// ---------------------------------------------------------------------------
__global__ __launch_bounds__(512) void fc12_kernel(
    const float* __restrict__ hfin,
    const float* __restrict__ Wd1, const float* __restrict__ bd1,
    const float* __restrict__ Wd2, const float* __restrict__ bd2,
    _Float16* __restrict__ h2out) {
  extern __shared__ __align__(16) char smem[];
  float* sH1 = (float*)smem;  // [64][350]
  const int tid = threadIdx.x;

  for (int i = tid; i < 64 * 350; i += 512) {
    int b = i / 350, j = i - b * 350;
    const float* wr = Wd1 + j * 64;
    const float* hb = hfin + b * 64;
    float s = bd1[j];
#pragma unroll 8
    for (int k = 0; k < 64; ++k) s += hb[k] * wr[k];
    sH1[i] = fmaxf(s, 0.f);
  }
  __syncthreads();

  for (int i = tid; i < 64 * 400; i += 512) {
    int b = i / 400, j = i - b * 400;
    const float* wr = Wd2 + j * 350;
    const float* hb = sH1 + b * 350;
    float s = bd2[j];
#pragma unroll 5
    for (int k = 0; k < 350; ++k) s += hb[k] * wr[k];
    h2out[b * 416 + j] = (_Float16)fmaxf(s, 0.f);
  }
  // zero K-padding 400..415
  for (int i = tid; i < 64 * 16; i += 512) {
    int b = i >> 4, j = 400 + (i & 15);
    h2out[b * 416 + j] = (_Float16)0.f;
  }
}

// ---------------------------------------------------------------------------
// fc3: out[64,100000] = h2 @ Wd3^T + bd3. HBM-bound: streams Wd3 (160 MB f32)
// exactly once per call (and it can stay resident in the 192 MB L2 across
// graph replays -> default RT temporal hint on purpose). Each block owns 256
// output columns; each wave owns one 16-col N-tile, so every Wd3 element is
// loaded by exactly one lane and its B fragment is reused over 4 M-tiles.
// Wd3 row-major [OUT][400] is exactly the W[n][k] layout fragB wants.
// ---------------------------------------------------------------------------
__global__ __launch_bounds__(512) void fc3_wmma_kernel(
    const _Float16* __restrict__ h2, const float* __restrict__ Wd3,
    const float* __restrict__ bd3, float* __restrict__ out) {
  extern __shared__ __align__(16) char smem[];
  _Float16* sA = (_Float16*)smem;  // [64][416] f16 activations
  const int tid = threadIdx.x;

  {  // stage activations
    const uint32_t* src = (const uint32_t*)h2;
    uint32_t* dst = (uint32_t*)sA;
    for (int i = tid; i < (64 * 416) / 2; i += 512) dst[i] = src[i];
  }
  __syncthreads();

  const int lane = tid & 31;
  const int wave = tid >> 5;
  const int hi8  = (lane >> 4) * 8;
  const int hi16 = (lane >> 4) * 16;
  const int n    = blockIdx.x * 256 + wave * 16 + (lane & 15);
  const int ncl  = n < 100000 ? n : 99999;  // keep loads in-bounds
  const float* wrow = Wd3 + (size_t)ncl * 400;

  v8f acc[4];
#pragma unroll
  for (int q = 0; q < 4; ++q) acc[q] = (v8f){};

  // 12 full K-chunks of 32 (K=0..383); convert f32 -> f16 in-flight.
  for (int kc = 0; kc < 12; ++kc) {
    const float* wp = wrow + kc * 32 + hi16;
    v8f lo = *(const v8f*)(wp);
    v8f hi = *(const v8f*)(wp + 8);
    AF f;
#pragma unroll
    for (int e = 0; e < 8; ++e) {
      f.v[e]     = (_Float16)lo[e];
      f.v[8 + e] = (_Float16)hi[e];
    }
#pragma unroll
    for (int mt = 0; mt < 4; ++mt) {
      v16h af = fragA(sA + (mt * 16 + (lane & 15)) * 416 + kc * 32, hi8);
      acc[mt] = WMMA_F16(af, f.v, acc[mt]);
    }
  }
  // Tail chunk: K=384..415; valid weights only for K<400 (upper lanes -> 0,
  // A side is zero-padded anyway). Divergence is loads-only; EXEC is full
  // again at the WMMA.
  {
    AF f;
    if (lane < 16) {
      v8f lo = *(const v8f*)(wrow + 384);
      v8f hi = *(const v8f*)(wrow + 392);
#pragma unroll
      for (int e = 0; e < 8; ++e) {
        f.v[e]     = (_Float16)lo[e];
        f.v[8 + e] = (_Float16)hi[e];
      }
    } else {
#pragma unroll
      for (int e = 0; e < 16; ++e) f.v[e] = (_Float16)0.f;
    }
#pragma unroll
    for (int mt = 0; mt < 4; ++mt) {
      v16h af = fragA(sA + (mt * 16 + (lane & 15)) * 416 + 384, hi8);
      acc[mt] = WMMA_F16(af, f.v, acc[mt]);
    }
  }

  if (n < 100000) {
    float bias = bd3[n];
#pragma unroll
    for (int mt = 0; mt < 4; ++mt)
#pragma unroll
      for (int r = 0; r < 8; ++r)
        out[(size_t)(mt * 16 + (lane >> 4) * 8 + r) * 100000 + n] =
            acc[mt][r] + bias;
  }
}

extern "C" void kernel_launch(void* const* d_in, const int* in_sizes, int n_in,
                              void* d_out, int out_size, void* d_ws,
                              size_t ws_size, hipStream_t stream) {
  const float* x    = (const float*)d_in[0];
  const float* dm   = (const float*)d_in[1];
  const float* fm   = (const float*)d_in[2];
  const float* Wih0 = (const float*)d_in[3];
  const float* Whh0 = (const float*)d_in[4];
  const float* b0   = (const float*)d_in[5];
  const float* Wih1 = (const float*)d_in[6];
  const float* Whh1 = (const float*)d_in[7];
  const float* b1   = (const float*)d_in[8];
  const float* Wd1  = (const float*)d_in[9];
  const float* bd1  = (const float*)d_in[10];
  const float* Wd2  = (const float*)d_in[11];
  const float* bd2  = (const float*)d_in[12];
  const float* Wd3  = (const float*)d_in[13];
  const float* bd3  = (const float*)d_in[14];
  float* out = (float*)d_out;

  char* ws = (char*)d_ws;
  float* hfin  = (float*)ws;                    // 64*64 f32
  _Float16* h2 = (_Float16*)(ws + 64 * 64 * 4); // 64*416 f16

  const size_t ldsLstm =
      (size_t)(256 * 32 + 3 * 256 * 64 + 2 * 64 * 32 + 2 * 64 * 64) * 2 +
      (size_t)(2 * 256 + 64 * 256) * 4;          // 206848 B (< 320 KB/WGP)
  const size_t ldsFc12 = (size_t)64 * 350 * 4;   // 89600 B
  const size_t ldsFc3  = (size_t)64 * 416 * 2;   // 53248 B

  hipFuncSetAttribute(reinterpret_cast<const void*>(lstm2_wmma_kernel),
                      hipFuncAttributeMaxDynamicSharedMemorySize, (int)ldsLstm);
  hipFuncSetAttribute(reinterpret_cast<const void*>(fc12_kernel),
                      hipFuncAttributeMaxDynamicSharedMemorySize, (int)ldsFc12);
  hipFuncSetAttribute(reinterpret_cast<const void*>(fc3_wmma_kernel),
                      hipFuncAttributeMaxDynamicSharedMemorySize, (int)ldsFc3);

  lstm2_wmma_kernel<<<1, 512, ldsLstm, stream>>>(x, dm, fm, Wih0, Whh0, b0,
                                                 Wih1, Whh1, b1, hfin);
  fc12_kernel<<<1, 512, ldsFc12, stream>>>(hfin, Wd1, bd1, Wd2, bd2, h2);
  fc3_wmma_kernel<<<(100000 + 255) / 256, 512, ldsFc3, stream>>>(h2, Wd3, bd3,
                                                                 out);
  (void)in_sizes; (void)n_in; (void)out_size; (void)ws_size;
}
